// DeConvRFNet_87540023427191
// MI455X (gfx1250) — matrix operand order repaired and checked by reading, not compiled, not run on hardware
//
#include <hip/hip_runtime.h>
#include <math.h>

typedef unsigned short u16;
typedef __attribute__((ext_vector_type(8)))  u16    u16x8;
typedef __attribute__((ext_vector_type(16))) u16    u16x16;
typedef __attribute__((ext_vector_type(16))) __bf16 v16bf;
typedef __attribute__((ext_vector_type(8)))  float  v8f;

struct P16 { u16x8 lo, hi; };

#define BDIM 4

__device__ __forceinline__ u16 f2bf(float f) {
  unsigned u = __float_as_uint(f);
  u += 0x7FFFu + ((u >> 16) & 1u);           // round-to-nearest-even
  return (u16)(u >> 16);
}

__device__ __forceinline__ size_t act_idx(int nchw, int C, int H, int W,
                                          int c, int b, int y, int x) {
  return nchw ? (((size_t)b * C + c) * H + y) * W + x
              : (((size_t)c * BDIM + b) * H + y) * W + x;
}

// ---------------- offset/modulation direct conv (small) ----------------
// out layout: [ch][b][Ho][Wo], ch in [0,3K): 0..2K-1 = offsets, 2K..3K-1 = 2*sigmoid(mod)
__global__ void k_offmod(const float* __restrict__ in, int nchw, int Cin, int Hi, int Wi,
                         int k, int stride, int pad, int Ho, int Wo,
                         const float* __restrict__ w_off, const float* __restrict__ b_off,
                         const float* __restrict__ w_mod, const float* __restrict__ b_mod,
                         float* __restrict__ om, int total)
{
  int tid = blockIdx.x * blockDim.x + threadIdx.x;
  if (tid >= total) return;
  int K = k * k;
  int x = tid % Wo; int t = tid / Wo;
  int y = t % Ho;   t /= Ho;
  int b = t % BDIM; int ch = t / BDIM;
  bool is_mod = (ch >= 2 * K);
  const float* wp = is_mod ? (w_mod + (size_t)(ch - 2 * K) * Cin * K)
                           : (w_off + (size_t)ch * Cin * K);
  float acc = is_mod ? b_mod[ch - 2 * K] : b_off[ch];
  int y0 = y * stride - pad, x0 = x * stride - pad;
  for (int c = 0; c < Cin; ++c)
    for (int ki = 0; ki < k; ++ki) {
      int yy = y0 + ki;
      if ((unsigned)yy >= (unsigned)Hi) continue;
      for (int kj = 0; kj < k; ++kj) {
        int xx = x0 + kj;
        if ((unsigned)xx >= (unsigned)Wi) continue;
        acc += in[act_idx(nchw, Cin, Hi, Wi, c, b, yy, xx)]
             * wp[(size_t)c * K + ki * k + kj];
      }
    }
  if (is_mod) acc = 2.f / (1.f + __expf(-acc));
  om[tid] = acc;      // tid decomposition == [ch][b][y][x]
}

// ---------------- modulated bilinear sampling -> bf16 im2col S_T[n][CKp] ----------------
__global__ void k_sample(const float* __restrict__ in, int nchw, int Cin, int Hi, int Wi,
                         int k, int stride, int pad, int Ho, int Wo,
                         const float* __restrict__ om, u16* __restrict__ S,
                         int CK, int CKp, long total)
{
  long tid = (long)blockIdx.x * blockDim.x + threadIdx.x;
  if (tid >= total) return;
  int  ck = (int)(tid % CKp);
  long n  = tid / CKp;                       // n = b*HW + y*Wo + x
  if (ck >= CK) { S[tid] = 0; return; }      // K-padding
  int K = k * k;
  int c = ck / K, kidx = ck % K;
  int HW = Ho * Wo;
  int x = (int)(n % Wo);
  int y = (int)((n / Wo) % Ho);
  int b = (int)(n / HW);
  size_t BHW = (size_t)BDIM * HW;
  float dy = om[(size_t)(2 * kidx    ) * BHW + n];
  float dx = om[(size_t)(2 * kidx + 1) * BHW + n];
  float mv = om[((size_t)2 * K + kidx) * BHW + n];
  int ki = kidx / k, kj = kidx % k;
  float py = dy + (float)(y * stride - pad + ki);
  float px = dx + (float)(x * stride - pad + kj);
  float y0f = floorf(py), x0f = floorf(px);
  int   yi = (int)y0f,    xi = (int)x0f;
  float wy = py - y0f,    wx = px - x0f;
#define G(yy, xx) ((((unsigned)(yy) < (unsigned)Hi) && ((unsigned)(xx) < (unsigned)Wi)) \
                   ? in[act_idx(nchw, Cin, Hi, Wi, c, b, (yy), (xx))] : 0.f)
  float v00 = G(yi,     xi);
  float v01 = G(yi,     xi + 1);
  float v10 = G(yi + 1, xi);
  float v11 = G(yi + 1, xi + 1);
#undef G
  float s = v00 * (1.f - wy) * (1.f - wx) + v01 * (1.f - wy) * wx
          + v10 * wy * (1.f - wx)         + v11 * wy * wx;
  S[tid] = f2bf(s * mv);
}

// ---------------- weight fp32 -> bf16, K-padded ----------------
__global__ void k_wcvt(const float* __restrict__ W, u16* __restrict__ Wb,
                       int CK, int CKp, int total)
{
  int tid = blockIdx.x * blockDim.x + threadIdx.x;
  if (tid >= total) return;
  int ck = tid % CKp, o = tid / CKp;
  Wb[tid] = (ck < CK) ? f2bf(W[(size_t)o * CK + ck]) : (u16)0;
}

// ---------------- WMMA GEMM: out[m][n] = sum_ck Wbf[m][ck] * S_T[n][ck] ----------------
// one wave computes a 16x64 stripe: 1 M-tile x 4 N-tiles. The A (weight) fragment is
// loaded once per k-step and fed to 4 v_wmma_f32_16x16x32_bf16 with independent
// accumulator chains (hides the 4-coexec bf16 WMMA RAW latency, 4x A-traffic reuse).
__global__ void __launch_bounds__(128)
k_gemm(const u16* __restrict__ A, const u16* __restrict__ Bm,
       float* __restrict__ out, int N, int CKp, int Ntiles)
{
  int lane = threadIdx.x & 31;
  int wave = threadIdx.x >> 5;
  int nt0  = (blockIdx.x * 4 + wave) * 4;    // 4 consecutive N-tiles per wave
  if (nt0 >= Ntiles) return;                 // wave-uniform (EXEC all-1s for WMMA)
  int mtile = blockIdx.y;
  int l16 = lane & 15;
  int hi  = lane >> 4;
  // A fragment: lane -> M = l16; half-wave selects K base {0,16}; contiguous 16 elems
  const u16* Arow = A  + (size_t)(mtile * 16 + l16) * CKp + hi * 16;
  // B fragments: lane -> N = l16; half-wave selects K base {0,16}; contiguous 16 elems
  const u16* B0 = Bm + (size_t)((nt0 + 0) * 16 + l16) * CKp + hi * 16;
  const u16* B1 = Bm + (size_t)((nt0 + 1) * 16 + l16) * CKp + hi * 16;
  const u16* B2 = Bm + (size_t)((nt0 + 2) * 16 + l16) * CKp + hi * 16;
  const u16* B3 = Bm + (size_t)((nt0 + 3) * 16 + l16) * CKp + hi * 16;
  v8f acc0 = {}, acc1 = {}, acc2 = {}, acc3 = {};
  for (int kk = 0; kk < CKp; kk += 32) {
    P16 ap;
    ap.lo = *(const u16x8*)(Arow + kk);
    ap.hi = *(const u16x8*)(Arow + kk + 8);
    v16bf av = __builtin_bit_cast(v16bf, ap);
    v16bf bv0 = __builtin_bit_cast(v16bf, *(const u16x16*)(B0 + kk));
    v16bf bv1 = __builtin_bit_cast(v16bf, *(const u16x16*)(B1 + kk));
    v16bf bv2 = __builtin_bit_cast(v16bf, *(const u16x16*)(B2 + kk));
    v16bf bv3 = __builtin_bit_cast(v16bf, *(const u16x16*)(B3 + kk));
    __builtin_prefetch(B0 + kk + 128, 0, 1);   // global_prefetch_b8
    __builtin_prefetch(B2 + kk + 128, 0, 1);
    acc0 = __builtin_amdgcn_wmma_f32_16x16x32_bf16(false, av, false, bv0,
                                                   (short)0, acc0, false, false);
    acc1 = __builtin_amdgcn_wmma_f32_16x16x32_bf16(false, av, false, bv1,
                                                   (short)0, acc1, false, false);
    acc2 = __builtin_amdgcn_wmma_f32_16x16x32_bf16(false, av, false, bv2,
                                                   (short)0, acc2, false, false);
    acc3 = __builtin_amdgcn_wmma_f32_16x16x32_bf16(false, av, false, bv3,
                                                   (short)0, acc3, false, false);
  }
  // C/D layout: VGPR v -> M = mbase+v, N = l16
  int mbase = mtile * 16 + hi * 8;
#pragma unroll
  for (int v = 0; v < 8; ++v) {
    size_t row = (size_t)(mbase + v) * N;
    out[row + (nt0 + 0) * 16 + l16] = acc0[v];
    out[row + (nt0 + 1) * 16 + l16] = acc1[v];
    out[row + (nt0 + 2) * 16 + l16] = acc2[v];
    out[row + (nt0 + 3) * 16 + l16] = acc3[v];
  }
}

// ---------------- batchnorm: per-channel stats (LDS reduce), then apply ----------------
__global__ void k_bn_stats(const float* __restrict__ x, float* __restrict__ st, int N)
{
  int c = blockIdx.x;
  const float* p = x + (size_t)c * N;
  float s = 0.f, q = 0.f;
  for (int i = threadIdx.x; i < N; i += blockDim.x) { float v = p[i]; s += v; q += v * v; }
  __shared__ float ss[256], qq[256];
  ss[threadIdx.x] = s; qq[threadIdx.x] = q;
  __syncthreads();
  for (int o = 128; o > 0; o >>= 1) {
    if ((int)threadIdx.x < o) {
      ss[threadIdx.x] += ss[threadIdx.x + o];
      qq[threadIdx.x] += qq[threadIdx.x + o];
    }
    __syncthreads();
  }
  if (threadIdx.x == 0) {
    float m = ss[0] / (float)N;
    float v = qq[0] / (float)N - m * m;
    st[2 * c]     = m;
    st[2 * c + 1] = rsqrtf(v + 1e-5f);
  }
}

__global__ void k_bn_apply(float* __restrict__ x, const float* __restrict__ st,
                           const float* __restrict__ g, const float* __restrict__ b,
                           int N, int relu, long total)
{
  long i = (long)blockIdx.x * blockDim.x + threadIdx.x;
  if (i >= total) return;
  int c = (int)(i / N);
  float v = (x[i] - st[2 * c]) * st[2 * c + 1] * g[c] + b[c];
  x[i] = relu ? fmaxf(v, 0.f) : v;
}

__global__ void k_add(float* __restrict__ h, const float* __restrict__ t, long total)
{
  long i = (long)blockIdx.x * blockDim.x + threadIdx.x;
  if (i < total) h[i] += t[i];
}

// ---------------- final 1x1 conv + relu -> NCHW (C=1 so flat index == n) ----------------
__global__ void k_final(const float* __restrict__ h, const float* __restrict__ w5,
                        const float* __restrict__ b5, float* __restrict__ out,
                        int N, int C)
{
  int n = blockIdx.x * blockDim.x + threadIdx.x;
  if (n >= N) return;
  float a = b5[0];
  for (int c = 0; c < C; ++c) a += h[(size_t)c * N + n] * w5[c];
  out[n] = fmaxf(a, 0.f);
}

// ---------------- host orchestration ----------------
static void run_dconv(const float* in, int nchw, int Cin, int Hi, int Wi,
                      int Cout, int k, int stride, int pad,
                      const float* w_off, const float* b_off,
                      const float* w_mod, const float* b_mod,
                      const float* wmain, float* out,
                      float* om, u16* Sb, u16* Wb, hipStream_t s)
{
  int Ho = (Hi + 2 * pad - k) / stride + 1;
  int Wo = (Wi + 2 * pad - k) / stride + 1;
  int K = k * k, CK = Cin * K, CKp = (CK + 31) & ~31;
  long N = (long)BDIM * Ho * Wo;
  {
    int total = 3 * K * BDIM * Ho * Wo;
    k_offmod<<<(total + 255) / 256, 256, 0, s>>>(in, nchw, Cin, Hi, Wi, k, stride, pad,
                                                 Ho, Wo, w_off, b_off, w_mod, b_mod, om, total);
  }
  {
    long total = N * CKp;
    k_sample<<<(int)((total + 255) / 256), 256, 0, s>>>(in, nchw, Cin, Hi, Wi, k, stride, pad,
                                                        Ho, Wo, om, Sb, CK, CKp, total);
  }
  {
    int total = Cout * CKp;
    k_wcvt<<<(total + 255) / 256, 256, 0, s>>>(wmain, Wb, CK, CKp, total);
  }
  {
    int Ntiles = (int)(N / 16);                 // 3136 or 784: both multiples of 16
    dim3 g((Ntiles + 15) / 16, Cout / 16);      // 4 waves * 4 N-tiles = 16 tiles per block
    k_gemm<<<g, 128, 0, s>>>(Wb, Sb, out, (int)N, CKp, Ntiles);
  }
}

static void run_bn(float* xb, int C, int N, const float* g, const float* b,
                   int relu, float* st, hipStream_t s)
{
  k_bn_stats<<<C, 256, 0, s>>>(xb, st, N);
  long total = (long)C * N;
  k_bn_apply<<<(int)((total + 255) / 256), 256, 0, s>>>(xb, st, g, b, N, relu, total);
}

extern "C" void kernel_launch(void* const* d_in, const int* in_sizes, int n_in,
                              void* d_out, int out_size, void* d_ws, size_t ws_size,
                              hipStream_t stream)
{
  (void)in_sizes; (void)n_in; (void)out_size; (void)ws_size;
#define F(i) ((const float*)d_in[(i)])
  // workspace carve (256B-aligned bump allocator)
  char* wsp = (char*)d_ws;
  auto carve = [&](size_t bytes) -> void* {
    void* p = (void*)wsp; wsp += (bytes + 255) & ~(size_t)255; return p;
  };
  float* h1 = (float*)carve((size_t)32  * 4 * 112 * 112 * 4);
  float* h2 = (float*)carve((size_t)64  * 4 * 56  * 56  * 4);
  float* hs = (float*)carve((size_t)128 * 4 * 56  * 56  * 4);
  float* t1 = (float*)carve((size_t)128 * 4 * 56  * 56  * 4);
  float* t2 = (float*)carve((size_t)128 * 4 * 56  * 56  * 4);
  float* om = (float*)carve((size_t)75  * 4 * 112 * 112 * 4);   // worst: conv1 (3K=75 ch)
  u16*   Sb = (u16*)  carve((size_t)12544 * 1152 * 2);          // worst: residual layers
  u16*   Wb = (u16*)  carve((size_t)128 * 1152 * 2);
  float* st = (float*)carve((size_t)128 * 2 * 4);

  // L1: 3 -> 32, k5 s2 p2, 224 -> 112 (input is NCHW)
  run_dconv(F(0), 1, 3, 224, 224, 32, 5, 2, 2, F(1), F(2), F(3), F(4), F(5),
            h1, om, Sb, Wb, stream);
  run_bn(h1, 32, 4 * 112 * 112, F(7), F(8), 1, st, stream);
  // L2: 32 -> 64, k3 s2 p1, 112 -> 56
  run_dconv(h1, 0, 32, 112, 112, 64, 3, 2, 1, F(9), F(10), F(11), F(12), F(13),
            h2, om, Sb, Wb, stream);
  run_bn(h2, 64, 4 * 56 * 56, F(15), F(16), 1, st, stream);
  // L3: 64 -> 128, k3 s1 p1
  run_dconv(h2, 0, 64, 56, 56, 128, 3, 1, 1, F(17), F(18), F(19), F(20), F(21),
            hs, om, Sb, Wb, stream);
  run_bn(hs, 128, 4 * 56 * 56, F(23), F(24), 1, st, stream);
  // 4x residual: shared weights r1a/r1b per reference
  long resN = (long)128 * 4 * 56 * 56;
  for (int r = 0; r < 4; ++r) {
    run_dconv(hs, 0, 128, 56, 56, 128, 3, 1, 1, F(25), F(26), F(27), F(28), F(29),
              t1, om, Sb, Wb, stream);
    run_bn(t1, 128, 4 * 56 * 56, F(31), F(32), 1, st, stream);
    run_dconv(t1, 0, 128, 56, 56, 128, 3, 1, 1, F(33), F(34), F(35), F(36), F(37),
              t2, om, Sb, Wb, stream);
    run_bn(t2, 128, 4 * 56 * 56, F(39), F(40), 0, st, stream);
    k_add<<<(int)((resN + 255) / 256), 256, 0, stream>>>(hs, t2, resN);
  }
  // L4: 128 -> 64
  run_dconv(hs, 0, 128, 56, 56, 64, 3, 1, 1, F(41), F(42), F(43), F(44), F(45),
            t1, om, Sb, Wb, stream);
  run_bn(t1, 64, 4 * 56 * 56, F(47), F(48), 1, st, stream);
  // final 1x1 conv + relu -> d_out (4,1,56,56)
  k_final<<<(12544 + 255) / 256, 256, 0, stream>>>(t1, F(49), F(50),
                                                   (float*)d_out, 12544, 64);
#undef F
}
